// RNNEncoderDecoderModel_86646670230113
// MI455X (gfx1250) — matrix operand order repaired and checked by reading, not compiled, not run on hardware
//
#include <hip/hip_runtime.h>
#include <hip/hip_bf16.h>
#include <stdint.h>
#include <math.h>

// Problem dims (fixed by the reference)
#define LL 2
#define BB 32
#define TT 64
#define HH 1024
#define EE 1024
#define VV 16000
#define G4H (4 * HH)

typedef __bf16 bf16_t;
typedef __attribute__((ext_vector_type(16))) __bf16 v16bf;
typedef __attribute__((ext_vector_type(8)))  float  v8f;

union Frag {
    v16bf v;
    uint4 q[2];
};

__device__ __forceinline__ float sigm(float x) { return 1.0f / (1.0f + __expf(-x)); }

// ---------------------------------------------------------------------------
// Fused dual-GEMM gate kernel:  g = A1 @ W1^T + A2 @ W2^T + bias
//   A1, A2 : [BB x HH] bf16 (h_prev and layer-input)
//   W1, W2 : [4H x HH] bf16 row-major (Whh, Wih)
//   g      : [BB x 4H] fp32
// One wave computes one 16x16 tile; K-loop of 32 with two WMMAs per iter.
// Fragment layouts follow CDNA5 ISA 7.12.2 (wave32):
//   A(16x32,bf16): lane m=L&15 holds K = 8*(L>>4)+0..7 and 16+8*(L>>4)+0..7
//   B(32x16,bf16): lane n=L&15 holds col n, K = 8*(L>>4)+0..7 and 16+8*(L>>4)+0..7
//   C(16x16,f32):  VGPR r -> row r + 8*(L>>4), col L&15
// ---------------------------------------------------------------------------
__global__ void __launch_bounds__(128)
lstm_gates_wmma(const bf16_t* __restrict__ A1, const bf16_t* __restrict__ W1,
                const bf16_t* __restrict__ A2, const bf16_t* __restrict__ W2,
                const float* __restrict__ bias, float* __restrict__ g)
{
    const int lane = threadIdx.x & 31;
    const int wave = threadIdx.x >> 5;
    const int nt   = blockIdx.x * 4 + wave;   // N tile (of 256)
    const int mt   = blockIdx.y;              // M tile (of 2)
    const int mrow = lane & 15;
    const int hi   = lane >> 4;
    const int ncol = nt * 16 + (lane & 15);

    const bf16_t* a1 = A1 + (size_t)(mt * 16 + mrow) * HH;
    const bf16_t* a2 = A2 + (size_t)(mt * 16 + mrow) * HH;
    const bf16_t* w1 = W1 + (size_t)ncol * HH;
    const bf16_t* w2 = W2 + (size_t)ncol * HH;

    v8f acc = {};
#pragma unroll 2
    for (int k0 = 0; k0 < HH; k0 += 32) {
        Frag fa1, fb1, fa2, fb2;
        fa1.q[0] = *(const uint4*)(a1 + k0 + 8 * hi);
        fa1.q[1] = *(const uint4*)(a1 + k0 + 16 + 8 * hi);
        fb1.q[0] = *(const uint4*)(w1 + k0 + 8 * hi);
        fb1.q[1] = *(const uint4*)(w1 + k0 + 16 + 8 * hi);
        fa2.q[0] = *(const uint4*)(a2 + k0 + 8 * hi);
        fa2.q[1] = *(const uint4*)(a2 + k0 + 16 + 8 * hi);
        fb2.q[0] = *(const uint4*)(w2 + k0 + 8 * hi);
        fb2.q[1] = *(const uint4*)(w2 + k0 + 16 + 8 * hi);
        acc = __builtin_amdgcn_wmma_f32_16x16x32_bf16(false, fa1.v, false, fb1.v,
                                                      (short)0, acc, false, false);
        acc = __builtin_amdgcn_wmma_f32_16x16x32_bf16(false, fa2.v, false, fb2.v,
                                                      (short)0, acc, false, false);
    }

    const float bv = bias[ncol];
    float* gout = g + (size_t)(mt * 16 + 8 * hi) * G4H + ncol;
#pragma unroll
    for (int r = 0; r < 8; ++r)
        gout[(size_t)r * G4H] = acc[r] + bv;
}

// ---------------------------------------------------------------------------
// Output projection: logits[:, t, :] = h_top @ Wout^T + bout
//   h_top [BB x HH] bf16, Wout [VV x HH] bf16, out [BB x TT x VV] fp32
// ---------------------------------------------------------------------------
__global__ void __launch_bounds__(128)
proj_wmma(const bf16_t* __restrict__ Atop, const bf16_t* __restrict__ W,
          const float* __restrict__ bout, float* __restrict__ out, int t)
{
    const int lane = threadIdx.x & 31;
    const int wave = threadIdx.x >> 5;
    const int nt   = blockIdx.x * 4 + wave;   // N tile (of 1000)
    const int mt   = blockIdx.y;              // M tile (of 2)
    const int mrow = lane & 15;
    const int hi   = lane >> 4;
    const int ncol = nt * 16 + (lane & 15);

    const bf16_t* a = Atop + (size_t)(mt * 16 + mrow) * HH;
    const bf16_t* w = W + (size_t)ncol * HH;

    v8f acc = {};
#pragma unroll 2
    for (int k0 = 0; k0 < HH; k0 += 32) {
        Frag fa, fb;
        fa.q[0] = *(const uint4*)(a + k0 + 8 * hi);
        fa.q[1] = *(const uint4*)(a + k0 + 16 + 8 * hi);
        fb.q[0] = *(const uint4*)(w + k0 + 8 * hi);
        fb.q[1] = *(const uint4*)(w + k0 + 16 + 8 * hi);
        acc = __builtin_amdgcn_wmma_f32_16x16x32_bf16(false, fa.v, false, fb.v,
                                                      (short)0, acc, false, false);
    }

    const float bv = bout[ncol];
#pragma unroll
    for (int r = 0; r < 8; ++r) {
        const int b = mt * 16 + r + 8 * hi;
        out[((size_t)b * TT + t) * VV + ncol] = acc[r] + bv;
    }
}

// ---------------------------------------------------------------------------
// Elementwise LSTM gate activation (in-place state update).
//   g [BB x 4H] f32; c [BB x HH] f32 (updated); h [BB x HH] bf16 (updated)
// ---------------------------------------------------------------------------
__global__ void lstm_act(const float* __restrict__ g, float* __restrict__ c,
                         bf16_t* __restrict__ h)
{
    int i = blockIdx.x * blockDim.x + threadIdx.x;
    if (i >= BB * HH) return;
    int b = i / HH, j = i - b * HH;
    const float* gb = g + (size_t)b * G4H;
    float iv = gb[j];
    float fv = gb[HH + j];
    float gv = gb[2 * HH + j];
    float ov = gb[3 * HH + j];
    float cn = sigm(fv) * c[i] + sigm(iv) * tanhf(gv);
    c[i] = cn;
    h[i] = (bf16_t)(sigm(ov) * tanhf(cn));
}

// ---------------------------------------------------------------------------
// Helpers: weight fp32->bf16 convert, bias combine, embedding gather, fills.
// ---------------------------------------------------------------------------
__global__ void cvt_f32_bf16(const float* __restrict__ src, bf16_t* __restrict__ dst, int n)
{
    int i = blockIdx.x * blockDim.x + threadIdx.x;
    if (i < n) dst[i] = (bf16_t)src[i];
}

__global__ void bias_combine(const float* __restrict__ a, const float* __restrict__ b,
                             float* __restrict__ o, int n)
{
    int i = blockIdx.x * blockDim.x + threadIdx.x;
    if (i < n) o[i] = a[i] + b[i];
}

// xb[(t*BB + b)*EE + e] = bf16(emb[ids[b*TT + t]*EE + e]),  t in [0, steps)
__global__ void gather_emb(const float* __restrict__ emb, const int* __restrict__ ids,
                           bf16_t* __restrict__ xb, int steps)
{
    int i = blockIdx.x * blockDim.x + threadIdx.x;
    int total = steps * BB * EE;
    if (i >= total) return;
    int e  = i % EE;
    int tb = i / EE;
    int b  = tb % BB;
    int t  = tb / BB;
    int id = ids[b * TT + t];
    xb[i] = (bf16_t)emb[(size_t)id * EE + e];
}

__global__ void fill_f32(float* __restrict__ p, float v, int n)
{
    int i = blockIdx.x * blockDim.x + threadIdx.x;
    if (i < n) p[i] = v;
}

__global__ void fill_bf16_zero(bf16_t* __restrict__ p, int n)
{
    int i = blockIdx.x * blockDim.x + threadIdx.x;
    if (i < n) p[i] = (bf16_t)0.0f;
}

// logits[:, 0, :] = 0
__global__ void zero_t0(float* __restrict__ out)
{
    int i = blockIdx.x * blockDim.x + threadIdx.x;
    if (i >= BB * VV) return;
    int b = i / VV, v = i - b * VV;
    out[((size_t)b * TT) * VV + v] = 0.0f;
}

// logits[b, 0, output_ids[j, 0]] = 1.0  for all b, j  (reference broadcast quirk)
__global__ void ones_t0(float* __restrict__ out, const int* __restrict__ out_ids)
{
    int b = threadIdx.x;  // 0..31
    int j = threadIdx.y;  // 0..31
    int id = out_ids[j * TT + 0];
    out[((size_t)b * TT) * VV + id] = 1.0f;
}

// ---------------------------------------------------------------------------
extern "C" void kernel_launch(void* const* d_in, const int* in_sizes, int n_in,
                              void* d_out, int out_size, void* d_ws, size_t ws_size,
                              hipStream_t stream)
{
    (void)in_sizes; (void)n_in; (void)out_size; (void)ws_size;

    const int*   input_ids  = (const int*)  d_in[0];
    const int*   output_ids = (const int*)  d_in[1];
    const float* enc_emb    = (const float*)d_in[2];
    const float* enc_Wih    = (const float*)d_in[3];
    const float* enc_Whh    = (const float*)d_in[4];
    const float* enc_bih    = (const float*)d_in[5];
    const float* enc_bhh    = (const float*)d_in[6];
    const float* dec_emb    = (const float*)d_in[7];
    const float* dec_Wih    = (const float*)d_in[8];
    const float* dec_Whh    = (const float*)d_in[9];
    const float* dec_bih    = (const float*)d_in[10];
    const float* dec_bhh    = (const float*)d_in[11];
    const float* Wout       = (const float*)d_in[12];
    const float* bout       = (const float*)d_in[13];
    float* out = (float*)d_out;

    // ---- workspace carve (bump allocator, 256B aligned) ----
    uint8_t* ws = (uint8_t*)d_ws;
    size_t off = 0;
    auto alloc = [&](size_t bytes) -> void* {
        void* p = ws + off;
        off = (off + bytes + 255) & ~(size_t)255;
        return p;
    };
    const size_t LW = (size_t)LL * G4H * HH;            // elems per 2-layer weight stack
    bf16_t* encWihB = (bf16_t*)alloc(LW * sizeof(bf16_t));
    bf16_t* encWhhB = (bf16_t*)alloc(LW * sizeof(bf16_t));
    bf16_t* decWihB = (bf16_t*)alloc(LW * sizeof(bf16_t));
    bf16_t* decWhhB = (bf16_t*)alloc(LW * sizeof(bf16_t));
    bf16_t* WoutB   = (bf16_t*)alloc((size_t)VV * HH * sizeof(bf16_t));
    float*  encBias = (float*) alloc((size_t)LL * G4H * sizeof(float));
    float*  decBias = (float*) alloc((size_t)LL * G4H * sizeof(float));
    bf16_t* enc_xb  = (bf16_t*)alloc((size_t)TT * BB * EE * sizeof(bf16_t));
    bf16_t* dec_xb  = (bf16_t*)alloc((size_t)(TT - 1) * BB * EE * sizeof(bf16_t));
    bf16_t* hbuf    = (bf16_t*)alloc((size_t)LL * BB * HH * sizeof(bf16_t));
    float*  cbuf    = (float*) alloc((size_t)LL * BB * HH * sizeof(float));
    float*  gbuf    = (float*) alloc((size_t)BB * G4H * sizeof(float));

    bf16_t* h0 = hbuf;
    bf16_t* h1 = hbuf + (size_t)BB * HH;
    float*  c0 = cbuf;
    float*  c1 = cbuf + (size_t)BB * HH;
    const size_t LS = (size_t)G4H * HH;   // per-layer weight stride (elems)

    const int TPB = 256;
    auto blocks = [](long n, int tpb) { return (int)((n + tpb - 1) / tpb); };

    // ---- one-time (per call) precompute: bf16 weights, fused biases, gathers ----
    cvt_f32_bf16<<<blocks(LW, TPB), TPB, 0, stream>>>(enc_Wih, encWihB, (int)LW);
    cvt_f32_bf16<<<blocks(LW, TPB), TPB, 0, stream>>>(enc_Whh, encWhhB, (int)LW);
    cvt_f32_bf16<<<blocks(LW, TPB), TPB, 0, stream>>>(dec_Wih, decWihB, (int)LW);
    cvt_f32_bf16<<<blocks(LW, TPB), TPB, 0, stream>>>(dec_Whh, decWhhB, (int)LW);
    cvt_f32_bf16<<<blocks((long)VV * HH, TPB), TPB, 0, stream>>>(Wout, WoutB, VV * HH);
    bias_combine<<<blocks(LL * G4H, TPB), TPB, 0, stream>>>(enc_bih, enc_bhh, encBias, LL * G4H);
    bias_combine<<<blocks(LL * G4H, TPB), TPB, 0, stream>>>(dec_bih, dec_bhh, decBias, LL * G4H);
    gather_emb<<<blocks((long)TT * BB * EE, TPB), TPB, 0, stream>>>(enc_emb, input_ids, enc_xb, TT);
    gather_emb<<<blocks((long)(TT - 1) * BB * EE, TPB), TPB, 0, stream>>>(dec_emb, output_ids, dec_xb, TT - 1);
    fill_bf16_zero<<<blocks((long)LL * BB * HH, TPB), TPB, 0, stream>>>(hbuf, LL * BB * HH);
    fill_f32<<<blocks((long)LL * BB * HH, TPB), TPB, 0, stream>>>(cbuf, 0.0f, LL * BB * HH);

    // logits[:, 0, :]
    zero_t0<<<blocks((long)BB * VV, TPB), TPB, 0, stream>>>(out);
    ones_t0<<<1, dim3(32, 32), 0, stream>>>(out, output_ids);

    const dim3 gGrid(G4H / 16 / 4, 2);   // (64, 2) waves-of-4 over 256 N tiles, 2 M tiles
    const dim3 gBlk(128);
    const dim3 pGrid(VV / 16 / 4, 2);    // (250, 2) over 1000 N tiles
    const int  aBlocks = blocks((long)BB * HH, TPB);

    // ---- encoder: 64 steps x 2 layers ----
    for (int t = 0; t < TT; ++t) {
        const bf16_t* xt = enc_xb + (size_t)t * BB * EE;
        lstm_gates_wmma<<<gGrid, gBlk, 0, stream>>>(h0, encWhhB, xt, encWihB, encBias, gbuf);
        lstm_act<<<aBlocks, TPB, 0, stream>>>(gbuf, c0, h0);
        lstm_gates_wmma<<<gGrid, gBlk, 0, stream>>>(h1, encWhhB + LS, h0, encWihB + LS,
                                                    encBias + G4H, gbuf);
        lstm_act<<<aBlocks, TPB, 0, stream>>>(gbuf, c1, h1);
    }

    // ---- decoder: 63 steps x 2 layers + vocab projection ----
    for (int s = 0; s < TT - 1; ++s) {
        const bf16_t* xt = dec_xb + (size_t)s * BB * EE;
        lstm_gates_wmma<<<gGrid, gBlk, 0, stream>>>(h0, decWhhB, xt, decWihB, decBias, gbuf);
        lstm_act<<<aBlocks, TPB, 0, stream>>>(gbuf, c0, h0);
        lstm_gates_wmma<<<gGrid, gBlk, 0, stream>>>(h1, decWhhB + LS, h0, decWihB + LS,
                                                    decBias + G4H, gbuf);
        lstm_act<<<aBlocks, TPB, 0, stream>>>(gbuf, c1, h1);
        proj_wmma<<<pGrid, gBlk, 0, stream>>>(h1, WoutB, bout, out, s + 1);
    }
}
